// SentiBert_78039555769029
// MI455X (gfx1250) — matrix operand order repaired and checked by reading, not compiled, not run on hardware
//
#include <hip/hip_runtime.h>
#include <hip/hip_bf16.h>

// ---------------- problem constants ----------------
constexpr int B_ = 64;
constexpr int S_ = 512;
constexpr int H_ = 768;
constexpr int F_ = 3072;
constexpr int C_ = 3;
constexpr int M_ = B_ * S_;           // 32768 rows

typedef __attribute__((ext_vector_type(16))) _Float16 v16h;
typedef __attribute__((ext_vector_type(8)))  _Float16 v8h;
typedef __attribute__((ext_vector_type(4)))  _Float16 v4h;
typedef __attribute__((ext_vector_type(8)))  float    v8f;

__device__ inline unsigned lds_off(const void* p) {
  // Low 32 bits of the flat shared-aperture address are the LDS byte offset.
  return (unsigned)(uintptr_t)p;
}

// ============================================================
// Prep kernels
// ============================================================
__global__ void convert_x_f16(const float* __restrict__ in,
                              _Float16* __restrict__ out, int n) {
  int i = (blockIdx.x * blockDim.x + threadIdx.x) * 4;
  if (i + 3 < n) {
    float4 v = *(const float4*)(in + i);
    v4h o = {(_Float16)v.x, (_Float16)v.y, (_Float16)v.z, (_Float16)v.w};
    *(v4h*)(out + i) = o;
  }
}

// WT[n][k] = (f16) W[k][n]   (W is [K][N] row-major)
__global__ void transpose_convert_f16(const float* __restrict__ W,
                                      _Float16* __restrict__ WT,
                                      int K, int N) {
  __shared__ float tile[32][33];
  int kb = blockIdx.y * 32, nb = blockIdx.x * 32;
  int tx = threadIdx.x, ty = threadIdx.y;
  for (int i = ty; i < 32; i += 8)
    tile[i][tx] = W[(size_t)(kb + i) * N + nb + tx];
  __syncthreads();
  for (int i = ty; i < 32; i += 8)
    WT[(size_t)(nb + i) * K + kb + tx] = (_Float16)tile[tx][i];
}

// ============================================================
// WMMA GEMM:  C[M,N] = A[M,K] * B[K,N] + bias,   B given as BT[N][K]
// f16 inputs, f32 accumulate, f16 output.
// Block tile 128x128, 8 waves (2 x 4), wave tile 64x32, K-step 32.
// Double-buffered LDS fed by async Global->LDS copies (ASYNCcnt).
// ============================================================
constexpr int LDSS = 40;   // halfs per LDS row (32 + 8 pad; 80B row stride)

__global__ __launch_bounds__(256) void gemm_xw_f16(
    const _Float16* __restrict__ A,   // [M,K]
    const _Float16* __restrict__ BT,  // [N,K]
    const float*    __restrict__ bias,// [N]
    _Float16* __restrict__ Cout,      // [M,N]
    int Ndim, int Kdim)
{
  __shared__ __align__(16) _Float16 As[2][128 * LDSS];
  __shared__ __align__(16) _Float16 Bs[2][128 * LDSS];

  const int tid  = threadIdx.x;
  const int lane = tid & 31;
  const int wid  = tid >> 5;
  const int wm   = wid & 1;           // 2 waves along M
  const int wn   = wid >> 1;          // 4 waves along N
  const int row0 = blockIdx.y * 128;
  const int col0 = blockIdx.x * 128;

  // Cooperative tile loader mapping: 2 threads per 32-half row,
  // each thread copies 32 bytes (2 x async b128) per matrix per stage.
  const int lr = tid >> 1;            // 0..127
  const int lo = (tid & 1) * 16;      // 0 or 16 halfs
  const _Float16* aG = A  + (size_t)(row0 + lr) * Kdim + lo;
  const _Float16* bG = BT + (size_t)(col0 + lr) * Kdim + lo;
  const unsigned aOff0 = lds_off(&As[0][lr * LDSS + lo]);
  const unsigned aOff1 = lds_off(&As[1][lr * LDSS + lo]);
  const unsigned bOff0 = lds_off(&Bs[0][lr * LDSS + lo]);
  const unsigned bOff1 = lds_off(&Bs[1][lr * LDSS + lo]);

  const int hsel = lane >> 4;         // K-half select per ISA VGPR layout
  const int rloc = lane & 15;         // row/col within 16

  v8f acc[4][2] = {};

  // Issue one stage of async Global->LDS copies (4 x b128 per thread).
  auto issue_stage = [&](int kb, int buf) {
    unsigned ao = buf ? aOff1 : aOff0;
    unsigned bo = buf ? bOff1 : bOff0;
    unsigned long long ga = (unsigned long long)(uintptr_t)(aG + kb);
    unsigned long long gb = (unsigned long long)(uintptr_t)(bG + kb);
    asm volatile(
        "global_load_async_to_lds_b128 %0, %1, off\n\t"
        "global_load_async_to_lds_b128 %0, %1, off offset:16\n\t"
        "global_load_async_to_lds_b128 %2, %3, off\n\t"
        "global_load_async_to_lds_b128 %2, %3, off offset:16"
        :: "v"(ao), "v"(ga), "v"(bo), "v"(gb) : "memory");
  };

  const int nk = Kdim >> 5;           // k-steps of 32
  issue_stage(0, 0);

  for (int i = 0; i < nk; i++) {
    const int buf = i & 1;
    if (i + 1 < nk) {
      issue_stage((i + 1) << 5, buf ^ 1);
      // Async loads complete in order: <=4 outstanding means stage i done.
      asm volatile("s_wait_asynccnt 0x4" ::: "memory");
    } else {
      asm volatile("s_wait_asynccnt 0x0" ::: "memory");
    }
    __syncthreads();   // all waves' copies for this buffer landed

    const _Float16* Ab = As[buf];
    const _Float16* Bb = Bs[buf];

    // B fragment: lane holds col n=rloc, k = e + 16*hsel (16 contiguous halfs)
    v16h bf[2];
#pragma unroll
    for (int nt = 0; nt < 2; nt++) {
      const _Float16* p = Bb + (wn * 32 + nt * 16 + rloc) * LDSS + (hsel << 4);
      v8h l = *(const v8h*)p;
      v8h hh = *(const v8h*)(p + 8);
      bf[nt] = __builtin_shufflevector(l, hh, 0,1,2,3,4,5,6,7,8,9,10,11,12,13,14,15);
    }
#pragma unroll
    for (int mt = 0; mt < 4; mt++) {
      // A fragment: lane holds row m=rloc, k = e + 16*hsel (16 contiguous halfs)
      const _Float16* p = Ab + (wm * 64 + mt * 16 + rloc) * LDSS + (hsel << 4);
      v8h l = *(const v8h*)p;
      v8h hh = *(const v8h*)(p + 8);
      v16h af = __builtin_shufflevector(l, hh, 0,1,2,3,4,5,6,7,8,9,10,11,12,13,14,15);
#pragma unroll
      for (int nt = 0; nt < 2; nt++)
        acc[mt][nt] = __builtin_amdgcn_wmma_f32_16x16x32_f16(
            false, af, false, bf[nt], (short)0, acc[mt][nt], false, false);
    }
    __syncthreads();   // done reading this buffer before it is overwritten
  }

  // Epilogue: C layout — lanes 0-15: M=r, lanes 16-31: M=r+8; N = rloc.
#pragma unroll
  for (int nt = 0; nt < 2; nt++) {
    int n = col0 + wn * 32 + nt * 16 + rloc;
    float bv = bias[n];
#pragma unroll
    for (int mt = 0; mt < 4; mt++) {
#pragma unroll
      for (int r = 0; r < 8; r++) {
        int m = row0 + wm * 64 + mt * 16 + r + (hsel << 3);
        Cout[(size_t)m * Ndim + n] = (_Float16)(acc[mt][nt][r] + bv);
      }
    }
  }
}

// ============================================================
// Reductions (wave32)
// ============================================================
__device__ inline float wave_sum(float v) {
#pragma unroll
  for (int o = 16; o > 0; o >>= 1) v += __shfl_down(v, o, 32);
  return v;
}
__device__ inline float block_sum(float v, float* red) {
  v = wave_sum(v);
  if ((threadIdx.x & 31) == 0) red[threadIdx.x >> 5] = v;
  __syncthreads();
  float s = 0.f;
#pragma unroll
  for (int i = 0; i < 8; i++) s += red[i];
  __syncthreads();
  return s;
}
__device__ inline float block_max(float v, float* red) {
#pragma unroll
  for (int o = 16; o > 0; o >>= 1) v = fmaxf(v, __shfl_down(v, o, 32));
  if ((threadIdx.x & 31) == 0) red[threadIdx.x >> 5] = v;
  __syncthreads();
  float s = red[0];
#pragma unroll
  for (int i = 1; i < 8; i++) s = fmaxf(s, red[i]);
  __syncthreads();
  return s;
}

// ============================================================
// Fused per-batch head: q(CLS) -> scores -> softmax -> ctx ->
// out-proj+LN1 -> FFN(gelu)+LN2 -> pooler tanh -> classifier softmax
// One block per batch, 256 threads (8 waves).
// ============================================================
__global__ __launch_bounds__(256) void fused_head(
    const float* __restrict__ X, const float* __restrict__ msk,
    const float* __restrict__ wq, const float* __restrict__ bq,
    const _Float16* __restrict__ Kf, const _Float16* __restrict__ Vf,
    const float* __restrict__ wo, const float* __restrict__ bo,
    const float* __restrict__ l1g, const float* __restrict__ l1b,
    const float* __restrict__ wi, const float* __restrict__ bi,
    const float* __restrict__ wo2, const float* __restrict__ bo2,
    const float* __restrict__ l2g, const float* __restrict__ l2b,
    const float* __restrict__ wp, const float* __restrict__ bp,
    const float* __restrict__ wc, const float* __restrict__ bc,
    float* __restrict__ out)
{
  const int b = blockIdx.x;
  const int tid = threadIdx.x;
  const int lane = tid & 31, w = tid >> 5;
  const int n0 = tid, n1 = tid + 256, n2 = tid + 512;

  __shared__ float xrow[H_];   // x[b,0,:] -> later pooled
  __shared__ float qv[H_];     // q -> later x1
  __shared__ float ctx1[H_];   // ctx -> later x2
  __shared__ float hbuf[F_];
  __shared__ float probs[S_];
  __shared__ float red[8];

  for (int i = tid; i < H_; i += 256) xrow[i] = X[(size_t)b * S_ * H_ + i];
  __syncthreads();

  // ---- q = x @ wq + bq (CLS row only) ----
  {
    float a0 = 0.f, a1 = 0.f, a2 = 0.f;
    for (int k = 0; k < H_; k++) {
      float xv = xrow[k]; const float* wr = wq + (size_t)k * H_;
      a0 += xv * wr[n0]; a1 += xv * wr[n1]; a2 += xv * wr[n2];
    }
    qv[n0] = a0 + bq[n0]; qv[n1] = a1 + bq[n1]; qv[n2] = a2 + bq[n2];
  }
  __syncthreads();

  // ---- scores[j] = q . K[b,j] / sqrt(768) + mask ----  (wave per key row)
  const float scale = 0.03608439182435161f;  // 1/sqrt(768)
  for (int j = w; j < S_; j += 8) {
    const _Float16* kr = Kf + ((size_t)(b * S_ + j)) * H_;
    float p = 0.f;
    for (int d = lane; d < H_; d += 32) p += qv[d] * (float)kr[d];
    p = wave_sum(p);
    if (lane == 0) probs[j] = p * scale + msk[(size_t)b * S_ + j];
  }
  __syncthreads();

  // ---- softmax over 512 ----
  {
    float s0 = probs[tid], s1 = probs[tid + 256];
    float mx = block_max(fmaxf(s0, s1), red);
    float e0 = __expf(s0 - mx), e1 = __expf(s1 - mx);
    float inv = 1.f / block_sum(e0 + e1, red);
    probs[tid] = e0 * inv; probs[tid + 256] = e1 * inv;
  }
  __syncthreads();

  // ---- ctx = probs @ V[b] ----
  {
    float c0 = 0.f, c1 = 0.f, c2 = 0.f;
    for (int j = 0; j < S_; j++) {
      float p = probs[j];
      const _Float16* vr = Vf + ((size_t)(b * S_ + j)) * H_;
      c0 += p * (float)vr[n0]; c1 += p * (float)vr[n1]; c2 += p * (float)vr[n2];
    }
    ctx1[n0] = c0; ctx1[n1] = c1; ctx1[n2] = c2;
  }
  __syncthreads();

  // ---- x1 = LN(ctx @ wo + bo + x) -> qv ----
  {
    float y0 = 0.f, y1 = 0.f, y2 = 0.f;
    for (int k = 0; k < H_; k++) {
      float cv = ctx1[k]; const float* wr = wo + (size_t)k * H_;
      y0 += cv * wr[n0]; y1 += cv * wr[n1]; y2 += cv * wr[n2];
    }
    y0 += bo[n0] + xrow[n0]; y1 += bo[n1] + xrow[n1]; y2 += bo[n2] + xrow[n2];
    float mu  = block_sum(y0 + y1 + y2, red) * (1.f / H_);
    float d0 = y0 - mu, d1 = y1 - mu, d2 = y2 - mu;
    float var = block_sum(d0 * d0 + d1 * d1 + d2 * d2, red) * (1.f / H_);
    float r = rsqrtf(var + 1e-12f);
    __syncthreads();                                   // qv reads done above
    qv[n0] = d0 * r * l1g[n0] + l1b[n0];
    qv[n1] = d1 * r * l1g[n1] + l1b[n1];
    qv[n2] = d2 * r * l1g[n2] + l1b[n2];
  }
  __syncthreads();

  // ---- h = gelu_exact(x1 @ wi + bi) ----
  {
    float a[12];
#pragma unroll
    for (int c = 0; c < 12; c++) a[c] = 0.f;
    for (int k = 0; k < H_; k++) {
      float xv = qv[k]; const float* wr = wi + (size_t)k * F_;
#pragma unroll
      for (int c = 0; c < 12; c++) a[c] += xv * wr[tid + c * 256];
    }
#pragma unroll
    for (int c = 0; c < 12; c++) {
      int n = tid + c * 256;
      float t = a[c] + bi[n];
      hbuf[n] = 0.5f * t * (1.f + erff(t * 0.7071067811865475f));
    }
  }
  __syncthreads();

  // ---- x2 = LN(h @ wo2 + bo2 + x1) -> ctx1 ----
  {
    float y0 = 0.f, y1 = 0.f, y2 = 0.f;
    for (int k = 0; k < F_; k++) {
      float hv = hbuf[k]; const float* wr = wo2 + (size_t)k * H_;
      y0 += hv * wr[n0]; y1 += hv * wr[n1]; y2 += hv * wr[n2];
    }
    y0 += bo2[n0] + qv[n0]; y1 += bo2[n1] + qv[n1]; y2 += bo2[n2] + qv[n2];
    float mu  = block_sum(y0 + y1 + y2, red) * (1.f / H_);
    float d0 = y0 - mu, d1 = y1 - mu, d2 = y2 - mu;
    float var = block_sum(d0 * d0 + d1 * d1 + d2 * d2, red) * (1.f / H_);
    float r = rsqrtf(var + 1e-12f);
    ctx1[n0] = d0 * r * l2g[n0] + l2b[n0];
    ctx1[n1] = d1 * r * l2g[n1] + l2b[n1];
    ctx1[n2] = d2 * r * l2g[n2] + l2b[n2];
  }
  __syncthreads();

  // ---- pooled = tanh(x2 @ wp + bp) -> xrow ----
  {
    float y0 = 0.f, y1 = 0.f, y2 = 0.f;
    for (int k = 0; k < H_; k++) {
      float xv = ctx1[k]; const float* wr = wp + (size_t)k * H_;
      y0 += xv * wr[n0]; y1 += xv * wr[n1]; y2 += xv * wr[n2];
    }
    __syncthreads();                                   // xrow no longer needed
    xrow[n0] = tanhf(y0 + bp[n0]);
    xrow[n1] = tanhf(y1 + bp[n1]);
    xrow[n2] = tanhf(y2 + bp[n2]);
  }
  __syncthreads();

  // ---- classifier + softmax(3) ----
  {
    float l0 = 0.f, l1 = 0.f, l2 = 0.f;
    for (int k = tid; k < H_; k += 256) {
      float pv = xrow[k]; const float* wr = wc + (size_t)k * C_;
      l0 += pv * wr[0]; l1 += pv * wr[1]; l2 += pv * wr[2];
    }
    l0 = block_sum(l0, red); l1 = block_sum(l1, red); l2 = block_sum(l2, red);
    if (tid == 0) {
      l0 += bc[0]; l1 += bc[1]; l2 += bc[2];
      float m = fmaxf(l0, fmaxf(l1, l2));
      float e0 = __expf(l0 - m), e1 = __expf(l1 - m), e2 = __expf(l2 - m);
      float is = 1.f / (e0 + e1 + e2);
      out[b * 3 + 0] = e0 * is; out[b * 3 + 1] = e1 * is; out[b * 3 + 2] = e2 * is;
    }
  }
}

// ============================================================
// Launcher
// ============================================================
extern "C" void kernel_launch(void* const* d_in, const int* in_sizes, int n_in,
                              void* d_out, int out_size, void* d_ws, size_t ws_size,
                              hipStream_t stream) {
  const float* X   = (const float*)d_in[0];
  const float* msk = (const float*)d_in[1];
  const float* wq  = (const float*)d_in[2];
  const float* bq  = (const float*)d_in[3];
  const float* wk  = (const float*)d_in[4];
  const float* bk  = (const float*)d_in[5];
  const float* wv  = (const float*)d_in[6];
  const float* bv  = (const float*)d_in[7];
  const float* wo  = (const float*)d_in[8];
  const float* bo  = (const float*)d_in[9];
  const float* l1g = (const float*)d_in[10];
  const float* l1b = (const float*)d_in[11];
  const float* wi  = (const float*)d_in[12];
  const float* bi  = (const float*)d_in[13];
  const float* wo2 = (const float*)d_in[14];
  const float* bo2 = (const float*)d_in[15];
  const float* l2g = (const float*)d_in[16];
  const float* l2b = (const float*)d_in[17];
  const float* wp  = (const float*)d_in[18];
  const float* bp  = (const float*)d_in[19];
  const float* wc  = (const float*)d_in[20];
  const float* bc  = (const float*)d_in[21];
  float* out = (float*)d_out;
  (void)in_sizes; (void)n_in; (void)out_size; (void)ws_size;

  char* ws = (char*)d_ws;
  size_t off = 0;
  auto alloc = [&](size_t bytes) -> void* {
    void* p = ws + off;
    off = (off + bytes + 255) & ~(size_t)255;
    return p;
  };
  _Float16* Xh  = (_Float16*)alloc((size_t)M_ * H_ * 2);   // 48 MB
  _Float16* WkT = (_Float16*)alloc((size_t)H_ * H_ * 2);
  _Float16* WvT = (_Float16*)alloc((size_t)H_ * H_ * 2);
  _Float16* Kf  = (_Float16*)alloc((size_t)M_ * H_ * 2);   // 48 MB
  _Float16* Vf  = (_Float16*)alloc((size_t)M_ * H_ * 2);   // 48 MB

  const int nx = M_ * H_;
  convert_x_f16<<<dim3(nx / 4 / 256), 256, 0, stream>>>(X, Xh, nx);
  dim3 tb(32, 8);
  transpose_convert_f16<<<dim3(H_ / 32, H_ / 32), tb, 0, stream>>>(wk, WkT, H_, H_);
  transpose_convert_f16<<<dim3(H_ / 32, H_ / 32), tb, 0, stream>>>(wv, WvT, H_, H_);

  // K and V projections: the only large GEMMs that the output depends on.
  gemm_xw_f16<<<dim3(H_ / 128, M_ / 128), 256, 0, stream>>>(Xh, WkT, bk, Kf, H_, H_);
  gemm_xw_f16<<<dim3(H_ / 128, M_ / 128), 256, 0, stream>>>(Xh, WvT, bv, Vf, H_, H_);

  fused_head<<<dim3(B_), 256, 0, stream>>>(X, msk, wq, bq, Kf, Vf, wo, bo,
                                           l1g, l1b, wi, bi, wo2, bo2,
                                           l2g, l2b, wp, bp, wc, bc, out);
}
